// LSTM_23759759081889
// MI455X (gfx1250) — compile-verified, matrix operand order checked by reading, not run on hardware
//
#include <hip/hip_runtime.h>
#include <hip/hip_bf16.h>

// ---------------------------------------------------------------------------
// LSTM on MI455X (gfx1250, wave32, WMMA).
//   B=64, T=512, D=1024, U=1024.
// Phase 1: preact[B*T,4096] = bf16(X)[B*T,1024] @ bf16(Wx)[1024,4096] + bias
//          (one big WMMA GEMM, compute-bound)
// Phase 2: ONE persistent kernel, 32 WGs; each WG keeps its 256 KB slice of
//          Wh resident in LDS (320 KB/WGP on CDNA5) and iterates all 512
//          timesteps with a device-wide atomic barrier per step. h chunks are
//          staged via global_load_async_to_lds_b128 (ASYNCcnt path).
// All matmuls use v_wmma_f32_16x16x32_bf16 (fp32 accumulate).
// ---------------------------------------------------------------------------

typedef __attribute__((ext_vector_type(16))) __bf16 v16bf;
typedef __attribute__((ext_vector_type(8)))  float  v8f;

union FragBF {
    v16bf v;
    unsigned int u[8];
};

__device__ __forceinline__ unsigned short f2bf(float f) {
    unsigned int u = __float_as_uint(f);
    unsigned int r = u + 0x7fffu + ((u >> 16) & 1u);  // RNE
    return (unsigned short)(r >> 16);
}

// ---------------------------------------------------------------------------
// Kernel: zero h ping-pong buffers + c state + global barrier counter
// ---------------------------------------------------------------------------
__global__ __launch_bounds__(256) void lstm_init(unsigned short* __restrict__ h0,
                                                 unsigned short* __restrict__ h1,
                                                 float* __restrict__ c,
                                                 unsigned int* __restrict__ bar) {
    int i = blockIdx.x * 256 + threadIdx.x;   // 0 .. 65535
    h0[i] = 0;
    h1[i] = 0;
    c[i]  = 0.0f;
    if (i == 0) *bar = 0u;
}

// ---------------------------------------------------------------------------
// Kernel: f32 -> bf16 conversion of inputs (33.5M elements, 4 per thread)
// ---------------------------------------------------------------------------
__global__ __launch_bounds__(256) void convert_x(const float* __restrict__ x,
                                                 unsigned short* __restrict__ xb) {
    int i = (blockIdx.x * 256 + threadIdx.x) * 4;
    float4 v = *(const float4*)(x + i);
    xb[i + 0] = f2bf(v.x);
    xb[i + 1] = f2bf(v.y);
    xb[i + 2] = f2bf(v.z);
    xb[i + 3] = f2bf(v.w);
}

// ---------------------------------------------------------------------------
// Kernel: pack weights into bf16 [K, 4096] (gate cols: u | i | f | o) + bias.
// ---------------------------------------------------------------------------
__global__ __launch_bounds__(256) void pack_w(
    const float* __restrict__ wxu, const float* __restrict__ wxi,
    const float* __restrict__ wxf, const float* __restrict__ wxo,
    const float* __restrict__ whu, const float* __restrict__ whi,
    const float* __restrict__ whf, const float* __restrict__ who,
    const float* __restrict__ bu,  const float* __restrict__ bi,
    const float* __restrict__ bf,  const float* __restrict__ bo,
    unsigned short* __restrict__ Wx, unsigned short* __restrict__ Wh,
    float* __restrict__ bias) {
    int idx = blockIdx.x * 256 + threadIdx.x;  // 0 .. 4194303
    int d   = idx >> 12;
    int col = idx & 4095;
    int g   = col >> 10;
    int u   = col & 1023;
    const float* wx = (g == 0) ? wxu : (g == 1) ? wxi : (g == 2) ? wxf : wxo;
    const float* wh = (g == 0) ? whu : (g == 1) ? whi : (g == 2) ? whf : who;
    Wx[idx] = f2bf(wx[d * 1024 + u]);
    Wh[idx] = f2bf(wh[d * 1024 + u]);
    if (idx < 4096) {
        const float* bb = (g == 0) ? bu : (g == 1) ? bi : (g == 2) ? bf : bo;
        bias[idx] = bb[u];
    }
}

// ---------------------------------------------------------------------------
// 16-bit WMMA operand fragment (ISA 7.12.2):
//   lane L: row/col = L%16, half = L/16
//   VGPR v holds K = (v/4)*16 + half*8 + (v%4)*2 (pair in one dword)
// ---------------------------------------------------------------------------
__device__ __forceinline__ void load_frag(FragBF& f, const unsigned short* lds,
                                          int rowbase, int half) {
#pragma unroll
    for (int v = 0; v < 8; ++v) {
        int k = ((v >> 2) << 4) + (half << 3) + ((v & 3) << 1);
        f.u[v] = *(const unsigned int*)(lds + rowbase + k);
    }
}

// ---------------------------------------------------------------------------
// Kernel: input-projection GEMM. C[32768,4096] = A @ Bm + bias (bf16->f32).
// Block tile 128x128, 8 waves x (2x4) 16x16 WMMA tiles.
// ---------------------------------------------------------------------------
__global__ __launch_bounds__(256) void gemm_xproj(
    const unsigned short* __restrict__ A,
    const unsigned short* __restrict__ Bm,
    const float* __restrict__ bias,
    float* __restrict__ C) {
    const int K = 1024, N = 4096;
    __shared__ unsigned short ldsA[128 * 32];  // [row][kk]
    __shared__ unsigned short ldsB[128 * 32];  // [col][kk] (transposed)

    const int row0 = blockIdx.y * 128;
    const int col0 = blockIdx.x * 128;
    const int tid  = threadIdx.x;
    const int lane = tid & 31;
    const int w    = tid >> 5;
    const int wm   = w & 3;
    const int wn   = w >> 2;
    const int l16  = lane & 15;
    const int hv   = lane >> 4;

    v8f acc[2][4] = {};

    for (int k0 = 0; k0 < K; k0 += 32) {
        {
            int idx = tid * 16;
            int r   = idx >> 5;
            int kk  = idx & 31;
            const uint4* src = (const uint4*)(A + (size_t)(row0 + r) * K + k0 + kk);
            uint4 d0 = src[0];
            uint4 d1 = src[1];
            *(uint4*)&ldsA[r * 32 + kk + 0] = d0;
            *(uint4*)&ldsA[r * 32 + kk + 8] = d1;
        }
#pragma unroll
        for (int rep = 0; rep < 2; ++rep) {
            int e  = tid * 8 + rep * 2048;
            int kk = e >> 7;
            int cc = e & 127;
            uint4 d = *(const uint4*)(Bm + (size_t)(k0 + kk) * N + col0 + cc);
            const unsigned short* p = (const unsigned short*)&d;
#pragma unroll
            for (int j = 0; j < 8; ++j) ldsB[(cc + j) * 32 + kk] = p[j];
        }
        if (k0 + 32 < K) {
            __builtin_prefetch(A + (size_t)(row0 + tid / 2) * K + k0 + 32, 0, 1);
            __builtin_prefetch(Bm + (size_t)(k0 + 32 + (tid & 31)) * N + col0, 0, 1);
        }
        __syncthreads();

        FragBF a[2], b[4];
#pragma unroll
        for (int mt = 0; mt < 2; ++mt)
            load_frag(a[mt], ldsA, (32 * wm + mt * 16 + l16) * 32, hv);
#pragma unroll
        for (int nt = 0; nt < 4; ++nt)
            load_frag(b[nt], ldsB, (64 * wn + nt * 16 + l16) * 32, hv);

#pragma unroll
        for (int mt = 0; mt < 2; ++mt)
#pragma unroll
            for (int nt = 0; nt < 4; ++nt)
                acc[mt][nt] = __builtin_amdgcn_wmma_f32_16x16x32_bf16(
                    false, a[mt].v, false, b[nt].v, (short)0, acc[mt][nt],
                    false, false);
        __syncthreads();
    }

#pragma unroll
    for (int mt = 0; mt < 2; ++mt) {
#pragma unroll
        for (int nt = 0; nt < 4; ++nt) {
            int n  = col0 + 64 * wn + nt * 16 + l16;
            int mb = row0 + 32 * wm + mt * 16 + hv * 8;
            float bv = bias[n];
#pragma unroll
            for (int g = 0; g < 8; ++g)
                C[(size_t)(mb + g) * N + n] = acc[mt][nt][g] + bv;
        }
    }
}

// ---------------------------------------------------------------------------
// Persistent recurrence kernel. 32 WGs x 256 threads; WG j owns U columns
// [32j, 32j+32) for all 4 gates. Wh slice (4x32x1024 bf16 = 256KB+pad) lives
// in LDS for the whole scan. Per step:
//   gates[64, 4x32] = h[64,1024] @ Wh_lds ; fused cell update ; h ping-pong;
//   device-wide barrier (monotonic atomic counter + s_sleep spin).
// h chunks staged with global_load_async_to_lds_b128 (ASYNCcnt).
// LDS strides padded (1028 / 264) to spread banks.
// ---------------------------------------------------------------------------
#define WSTRIDE 1028   // elements per (gate,col) row of resident Wh slice
#define HSTRIDE 264    // elements per row of h staging tile (64 x 256 chunk)
#define LDSW_BYTES (128 * WSTRIDE * 2)        // 263168
#define LDSH_BYTES (64 * HSTRIDE * 2)         // 33792 (aliases gate buffer)

__global__ __launch_bounds__(256) void lstm_scan(
    unsigned short* __restrict__ h0,          // [64,1024] bf16 ping
    unsigned short* __restrict__ h1,          // [64,1024] bf16 pong
    const unsigned short* __restrict__ Wh,    // [1024,4096] bf16
    const float* __restrict__ preact,         // [B*T,4096] f32
    float* __restrict__ cbuf,                 // [64,1024] f32
    float* __restrict__ out,                  // [B,T,U] f32
    unsigned int* __restrict__ bar) {
    const int U = 1024, N4 = 4096, T = 512;
    extern __shared__ char smem[];
    unsigned short* ldsW = (unsigned short*)smem;                 // [128][WSTRIDE]
    unsigned short* ldsH = (unsigned short*)(smem + LDSW_BYTES);  // [64][HSTRIDE]
    float*          gbuf = (float*)(smem + LDSW_BYTES);           // alias: [4][64][32]

    const int col0 = blockIdx.x * 32;
    const int tid  = threadIdx.x;
    const int lane = tid & 31;
    const int w    = tid >> 5;
    const int g    = w >> 1;      // gate owned by this wave
    const int nh   = w & 1;       // 16-col tile within the 32 cols
    const int l16  = lane & 15;
    const int hv   = lane >> 4;

    // ---- prologue: park Wh slice in LDS, transposed to [gate*32+col][K] ---
#pragma unroll 4
    for (int rep = 0; rep < 64; ++rep) {
        int idx = (rep * 256 + tid) * 8;      // 0 .. 131071
        int k   = idx >> 7;                   // 0 .. 1023
        int cc  = idx & 127;
        int g2  = cc >> 5;
        int c   = cc & 31;                    // multiple of 8
        uint4 d = *(const uint4*)(Wh + (size_t)k * N4 + g2 * 1024 + col0 + c);
        const unsigned short* p = (const unsigned short*)&d;
#pragma unroll
        for (int j = 0; j < 8; ++j)
            ldsW[(g2 * 32 + c + j) * WSTRIDE + k] = p[j];
    }

    // ---- scan over time ---------------------------------------------------
    for (int t = 0; t < T; ++t) {
        const unsigned short* hin = (t & 1) ? h1 : h0;
        unsigned short*       hout = (t & 1) ? h0 : h1;

        v8f acc[4] = {};   // 4 M-tiles x one 16-col N-tile (this wave's gate)

        for (int k0 = 0; k0 < U; k0 += 256) {
            // stage h[64, k0:k0+256] -> LDS via async copies (64 elems/thread)
#pragma unroll
            for (int q = 0; q < 8; ++q) {
                int idx = (q * 256 + tid) * 8;     // 0 .. 16383
                int r   = idx >> 8;
                int kk  = idx & 255;               // multiple of 8
                const unsigned short* gp = hin + r * U + k0 + kk;
                unsigned int lds_off =
                    (unsigned int)(unsigned long long)&ldsH[r * HSTRIDE + kk];
                asm volatile("global_load_async_to_lds_b128 %0, %1, off"
                             :: "v"(lds_off), "v"(gp) : "memory");
            }
            asm volatile("s_wait_asynccnt 0x0" ::: "memory");
            __syncthreads();

#pragma unroll
            for (int kk2 = 0; kk2 < 8; ++kk2) {
                FragBF a[4], b;
                load_frag(b, ldsW,
                          (g * 32 + nh * 16 + l16) * WSTRIDE + k0 + kk2 * 32, hv);
#pragma unroll
                for (int mt = 0; mt < 4; ++mt)
                    load_frag(a[mt], ldsH, (mt * 16 + l16) * HSTRIDE + kk2 * 32, hv);
#pragma unroll
                for (int mt = 0; mt < 4; ++mt)
                    acc[mt] = __builtin_amdgcn_wmma_f32_16x16x32_bf16(
                        false, a[mt].v, false, b.v, (short)0, acc[mt],
                        false, false);
            }
            __syncthreads();   // protect ldsH restage / gbuf alias
        }

        // spill gate preacts to LDS (gbuf aliases ldsH; barrier above covers)
#pragma unroll
        for (int mt = 0; mt < 4; ++mt) {
            int nl = nh * 16 + l16;
            int mb = mt * 16 + hv * 8;
#pragma unroll
            for (int gg = 0; gg < 8; ++gg)
                gbuf[g * 2048 + (mb + gg) * 32 + nl] = acc[mt][gg];
        }
        __syncthreads();

        // fused elementwise cell update (64 rows x 32 cols, 8 per thread)
        for (int i = tid; i < 2048; i += 256) {
            int b2 = i >> 5;
            int cu = i & 31;
            int u  = col0 + cu;
            const float* pr = preact + (size_t)(b2 * T + t) * N4;
            float pu = gbuf[0 * 2048 + b2 * 32 + cu] + pr[u];
            float pi = gbuf[1 * 2048 + b2 * 32 + cu] + pr[1024 + u];
            float pf = gbuf[2 * 2048 + b2 * 32 + cu] + pr[2048 + u];
            float po = gbuf[3 * 2048 + b2 * 32 + cu] + pr[3072 + u];
            float uu = tanhf(pu);
            float ii = 1.0f / (1.0f + __expf(-pi));
            float ff = 1.0f / (1.0f + __expf(-pf));
            float oo = 1.0f / (1.0f + __expf(-po));
            float cp = cbuf[b2 * U + u];
            float cn = ii * uu + ff * cp;
            float hn = oo * tanhf(cn);
            cbuf[b2 * U + u] = cn;
            out[((size_t)b2 * T + t) * U + u] = hn;
            hout[b2 * U + u] = f2bf(hn);
        }

        // device-wide barrier: monotonic counter, target = 32*(t+1)
        __threadfence();
        __syncthreads();
        if (tid == 0) {
            atomicAdd(bar, 1u);
            unsigned target = 32u * (unsigned)(t + 1);
            while (atomicAdd(bar, 0u) < target) {
                __builtin_amdgcn_s_sleep(1);
            }
        }
        __syncthreads();
        __threadfence();
    }
}

// ---------------------------------------------------------------------------
// Host launcher (graph-capturable: launches only, on `stream`).
// Workspace (~621 MB): Xbf16 | Wx | Wh | bias | preact | h0 | h1 | c | bar
// ---------------------------------------------------------------------------
extern "C" void kernel_launch(void* const* d_in, const int* in_sizes, int n_in,
                              void* d_out, int out_size, void* d_ws, size_t ws_size,
                              hipStream_t stream) {
    (void)in_sizes; (void)n_in; (void)out_size; (void)ws_size;
    const int B = 64, T = 512, D = 1024, U = 1024;

    const float* x   = (const float*)d_in[0];
    const float* wxu = (const float*)d_in[1];
    const float* whu = (const float*)d_in[2];
    const float* bu  = (const float*)d_in[3];
    const float* wxi = (const float*)d_in[4];
    const float* whi = (const float*)d_in[5];
    const float* bi  = (const float*)d_in[6];
    const float* wxf = (const float*)d_in[7];
    const float* whf = (const float*)d_in[8];
    const float* bf  = (const float*)d_in[9];
    const float* wxo = (const float*)d_in[10];
    const float* who = (const float*)d_in[11];
    const float* bo  = (const float*)d_in[12];
    float* out = (float*)d_out;

    char* ws = (char*)d_ws;
    size_t off = 0;
    unsigned short* Xb   = (unsigned short*)(ws + off); off += (size_t)B * T * D * 2;
    unsigned short* Wx   = (unsigned short*)(ws + off); off += (size_t)D * 4 * U * 2;
    unsigned short* Wh   = (unsigned short*)(ws + off); off += (size_t)U * 4 * U * 2;
    float*          bias = (float*)(ws + off);          off += (size_t)4 * U * 4;
    float*          pre  = (float*)(ws + off);          off += (size_t)B * T * 4 * U * 4;
    unsigned short* h0   = (unsigned short*)(ws + off); off += (size_t)B * U * 2;
    unsigned short* h1   = (unsigned short*)(ws + off); off += (size_t)B * U * 2;
    float*          cst  = (float*)(ws + off);          off += (size_t)B * U * 4;
    unsigned int*   bar  = (unsigned int*)(ws + off);   off += 256;

    // 1) zero h/c state + barrier counter
    lstm_init<<<(B * U) / 256, 256, 0, stream>>>(h0, h1, cst, bar);

    // 2) inputs -> bf16
    convert_x<<<(B * T * D) / (256 * 4), 256, 0, stream>>>(x, Xb);

    // 3) pack weights + bias
    pack_w<<<(D * 4 * U) / 256, 256, 0, stream>>>(wxu, wxi, wxf, wxo,
                                                  whu, whi, whf, who,
                                                  bu, bi, bf, bo, Wx, Wh, bias);

    // 4) input projections: preact = Xb @ Wx + bias
    dim3 g1(4 * U / 128, B * T / 128);
    gemm_xproj<<<g1, 256, 0, stream>>>(Xb, Wx, bias, pre);

    // 5) whole recurrence in one persistent kernel (Wh resident in LDS)
    size_t smem = (size_t)(LDSW_BYTES + LDSH_BYTES);
    lstm_scan<<<32, 256, smem, stream>>>(h0, h1, Wh, pre, cst, out, bar);
}
